// ScaledDotProductAttention_38852274159621
// MI455X (gfx1250) — matrix operand order repaired and checked by reading, not compiled, not run on hardware
//
#include <hip/hip_runtime.h>

typedef __attribute__((ext_vector_type(16))) _Float16 v16h;
typedef __attribute__((ext_vector_type(8)))  _Float16 v8h;
typedef __attribute__((ext_vector_type(8)))  float    v8f;
typedef __attribute__((ext_vector_type(4)))  float    v4f;

#define S_LEN   2048
#define D_HEAD  64
#define QB      128   // query rows per workgroup
#define KT      32    // key columns per inner iteration
#define NTHREADS 256  // 8 wave32s

// WMMA A-fragment (16x32 f16): lane<16 holds K {0..7,16..23}, lane>=16 holds
// K {8..15,24..31} of its row. sel = 0 or 8 halves.
__device__ __forceinline__ v16h load_fragA(const _Float16* row, int sel) {
  v8h lo = *(const v8h*)(row + sel);
  v8h hi = *(const v8h*)(row + sel + 16);
  v16h r;
#pragma unroll
  for (int i = 0; i < 8; ++i) { r[i] = lo[i]; r[i + 8] = hi[i]; }
  return r;
}

// WMMA B-fragment (32x16 f16): lane<16 holds K {0..7,16..23}, lane>=16 holds
// K {8..15,24..31} of its column (stored as a contiguous row here). selB = 0 or 8 halves.
__device__ __forceinline__ v16h load_fragB(const _Float16* row, int selB) {
  v8h lo = *(const v8h*)(row + selB);
  v8h hi = *(const v8h*)(row + selB + 16);
  v16h r;
#pragma unroll
  for (int i = 0; i < 8; ++i) { r[i] = lo[i]; r[i + 8] = hi[i]; }
  return r;
}

__device__ __forceinline__ v8f wmma16(v16h a, v16h b, v8f c) {
  return __builtin_amdgcn_wmma_f32_16x16x32_f16(
      /*neg_a=*/false, a, /*neg_b=*/false, b,
      /*c_mod=*/(short)0, c, /*reuse_a=*/false, /*reuse_b=*/false);
}

// pack 8 fp32 (two v4f) into one v8h
__device__ __forceinline__ v8h pack8(v4f a, v4f b) {
  v8h h;
#pragma unroll
  for (int j = 0; j < 4; ++j) {
    h[j]     = (_Float16)a[j];
    h[j + 4] = (_Float16)b[j];
  }
  return h;
}

__global__ __launch_bounds__(NTHREADS) void attn_fwd_causal(
    const float* __restrict__ Qg_, const float* __restrict__ Kg_,
    const float* __restrict__ Vg_, float* __restrict__ Og_,
    float* __restrict__ Pg_) {
  __shared__ alignas(16) _Float16 sQ[QB][D_HEAD];        // 16 KB
  __shared__ alignas(16) _Float16 sK[2][KT][D_HEAD];     // 8 KB (double buf)
  __shared__ alignas(16) _Float16 sVt[2][D_HEAD][KT];    // 8 KB (V^T, double buf)

  const int nqb = S_LEN / QB;
  const int bh  = blockIdx.x / nqb;
  const int qb  = (blockIdx.x % nqb) * QB;

  const float* Qg = Qg_ + (size_t)bh * S_LEN * D_HEAD;
  const float* Kg = Kg_ + (size_t)bh * S_LEN * D_HEAD;
  const float* Vg = Vg_ + (size_t)bh * S_LEN * D_HEAD;
  float* Og = Og_ + (size_t)bh * S_LEN * D_HEAD;
  float* Pg = Pg_ + (size_t)bh * S_LEN * S_LEN;

  const int tid   = threadIdx.x;
  const int w     = tid >> 5;          // wave id (warpSize==32)
  const int lane  = tid & 31;
  const int lrow  = lane & 15;
  const int sel   = (lane & 16) ? 8 : 0;   // A-frag half-select
  const int selB  = (lane & 16) ? 8 : 0;   // B-frag half-select
  const int qglob = qb + w * 16 + lrow;    // this lane's query column
  const int qmin  = qb + w * 16;           // wave's minimum query row

  // per-thread staging coordinates: 8 contiguous floats of a K/V tile row
  const int r8 = tid >> 3;             // key row within tile (0..31)
  const int c8 = (tid & 7) * 8;        // column (0,8,...,56)
  const size_t goff = (size_t)r8 * D_HEAD + c8;

  // ---- stage Q block (pre-scaled by 1/sqrt(D)=0.125) as f16, b128 I/O ----
  for (int i0 = tid * 8; i0 < QB * D_HEAD; i0 += NTHREADS * 8) {
    int r = i0 >> 6, c = i0 & 63;
    const float* p = Qg + (size_t)(qb + r) * D_HEAD + c;
    v4f a = *(const v4f*)p, b = *(const v4f*)(p + 4);
    v8h h;
#pragma unroll
    for (int j = 0; j < 4; ++j) {
      h[j]     = (_Float16)(a[j] * 0.125f);
      h[j + 4] = (_Float16)(b[j] * 0.125f);
    }
    *(v8h*)&sQ[r][c] = h;
  }

  const int nkt = qb / KT + QB / KT;   // key tiles needed (causal)
  float m = -3.0e38f, l = 0.0f;

  // ================= PASS 1: softmax stats (online) =================
  // prologue: stage K tile 0 into buffer 0
  {
    v4f a = *(const v4f*)(Kg + goff), b = *(const v4f*)(Kg + goff + 4);
    *(v8h*)&sK[0][r8][c8] = pack8(a, b);
  }
  __syncthreads();

  const v16h bq0 = load_fragB(&sQ[w * 16 + lrow][0],  selB);
  const v16h bq1 = load_fragB(&sQ[w * 16 + lrow][32], selB);

  for (int jt = 0; jt < nkt; ++jt) {
    const int cur = jt & 1;
    const bool have_next = (jt + 1 < nkt);

    // issue next tile's global loads (latency overlapped with WMMA below)
    v4f ka, kb;
    if (have_next) {
      const float* p = Kg + (size_t)(jt + 1) * KT * D_HEAD + goff;
      ka = *(const v4f*)p;
      kb = *(const v4f*)(p + 4);
    }

    // S^T tiles: A = K rows (keys x D-chunk), B = Q^T  => D = K.Q^T
    v8f c0 = {}, c1 = {};
    c0 = wmma16(load_fragA(&sK[cur][lrow][0],       sel), bq0, c0);
    c0 = wmma16(load_fragA(&sK[cur][lrow][32],      sel), bq1, c0);
    c1 = wmma16(load_fragA(&sK[cur][16 + lrow][0],  sel), bq0, c1);
    c1 = wmma16(load_fragA(&sK[cur][16 + lrow][32], sel), bq1, c1);

    const int kb0 = jt * KT + sel;           // key index of c0[0] for this lane
    float s0[8], s1[8];
    if (jt * KT + (KT - 1) <= qmin) {
      // fully below diagonal for every lane of this wave: no masking (common)
#pragma unroll
      for (int r = 0; r < 8; ++r) { s0[r] = c0[r]; s1[r] = c1[r]; }
    } else {
#pragma unroll
      for (int r = 0; r < 8; ++r) {
        s0[r] = (kb0 + r      > qglob) ? -1.0e9f : c0[r];
        s1[r] = (kb0 + 16 + r > qglob) ? -1.0e9f : c1[r];
      }
    }
    float tmax = -3.0e38f;
#pragma unroll
    for (int r = 0; r < 8; ++r) tmax = fmaxf(tmax, fmaxf(s0[r], s1[r]));
    tmax = fmaxf(tmax, __shfl_xor(tmax, 16));   // pair lanes hold same query
    float mn   = fmaxf(m, tmax);
    float corr = __expf(m - mn);
    float ls = 0.0f;
#pragma unroll
    for (int r = 0; r < 8; ++r) ls += __expf(s0[r] - mn) + __expf(s1[r] - mn);
    l = l * corr + ls;
    m = mn;

    // spill prefetched tile into the other buffer, then one barrier
    if (have_next) {
      *(v8h*)&sK[1 - cur][r8][c8] = pack8(ka, kb);
    }
    __syncthreads();
  }
  l += __shfl_xor(l, 16);                        // combine the two key halves
  const float linv = 1.0f / l;

  // ================= PASS 2: P = softmax, O += P.V =================
  v8f acc0 = {}, acc1 = {}, acc2 = {}, acc3 = {};

  // prologue: stage K/V tile 0 into buffer 0
  {
    v4f a = *(const v4f*)(Kg + goff), b = *(const v4f*)(Kg + goff + 4);
    *(v8h*)&sK[0][r8][c8] = pack8(a, b);
    v4f va = *(const v4f*)(Vg + goff), vb = *(const v4f*)(Vg + goff + 4);
#pragma unroll
    for (int j = 0; j < 4; ++j) {
      sVt[0][c8 + j][r8]     = (_Float16)va[j];
      sVt[0][c8 + 4 + j][r8] = (_Float16)vb[j];
    }
  }
  __syncthreads();

  for (int jt = 0; jt < nkt; ++jt) {
    const int cur = jt & 1;
    const bool have_next = (jt + 1 < nkt);

    v4f ka, kb, va, vb;
    if (have_next) {
      const float* pk = Kg + (size_t)(jt + 1) * KT * D_HEAD + goff;
      const float* pv = Vg + (size_t)(jt + 1) * KT * D_HEAD + goff;
      ka = *(const v4f*)pk;  kb = *(const v4f*)(pk + 4);
      va = *(const v4f*)pv;  vb = *(const v4f*)(pv + 4);
    }

    v8f c0 = {}, c1 = {};
    c0 = wmma16(load_fragA(&sK[cur][lrow][0],       sel), bq0, c0);
    c0 = wmma16(load_fragA(&sK[cur][lrow][32],      sel), bq1, c0);
    c1 = wmma16(load_fragA(&sK[cur][16 + lrow][0],  sel), bq0, c1);
    c1 = wmma16(load_fragA(&sK[cur][16 + lrow][32], sel), bq1, c1);

    const int kb0 = jt * KT + sel;
    float s0[8], s1[8];
    if (jt * KT + (KT - 1) <= qmin) {
#pragma unroll
      for (int r = 0; r < 8; ++r) { s0[r] = c0[r]; s1[r] = c1[r]; }
    } else {
#pragma unroll
      for (int r = 0; r < 8; ++r) {
        s0[r] = (kb0 + r      > qglob) ? -1.0e9f : c0[r];
        s1[r] = (kb0 + 16 + r > qglob) ? -1.0e9f : c1[r];
      }
    }

    v16h ap = {};                   // P as WMMA A-fragment (layout match: free)
    v4f p0lo, p0hi, p1lo, p1hi;
#pragma unroll
    for (int r = 0; r < 8; ++r) {
      float p0 = __expf(s0[r] - m) * linv;  // masked -> exp underflow -> 0
      float p1 = __expf(s1[r] - m) * linv;
      ap[r]     = (_Float16)p0;
      ap[r + 8] = (_Float16)p1;
      if (r < 4) { p0lo[r] = p0;     p1lo[r] = p1;     }
      else       { p0hi[r - 4] = p0; p1hi[r - 4] = p1; }
    }
    // p_attn is a 1 GB write-once stream: keep it out of L2 (non-temporal)
    float* prow = Pg + (size_t)qglob * S_LEN + kb0;
    __builtin_nontemporal_store(p0lo, (v4f*)(prow));
    __builtin_nontemporal_store(p0hi, (v4f*)(prow + 4));
    __builtin_nontemporal_store(p1lo, (v4f*)(prow + 16));
    __builtin_nontemporal_store(p1hi, (v4f*)(prow + 20));

    // O += P(16q x 32k) . V(32k x 16d), 4 d-tiles
    acc0 = wmma16(ap, load_fragB(&sVt[cur][ 0 + lrow][0], selB), acc0);
    acc1 = wmma16(ap, load_fragB(&sVt[cur][16 + lrow][0], selB), acc1);
    acc2 = wmma16(ap, load_fragB(&sVt[cur][32 + lrow][0], selB), acc2);
    acc3 = wmma16(ap, load_fragB(&sVt[cur][48 + lrow][0], selB), acc3);

    if (have_next) {
      *(v8h*)&sK[1 - cur][r8][c8] = pack8(ka, kb);
#pragma unroll
      for (int j = 0; j < 4; ++j) {
        sVt[1 - cur][c8 + j][r8]     = (_Float16)va[j];
        sVt[1 - cur][c8 + 4 + j][r8] = (_Float16)vb[j];
      }
    }
    __syncthreads();
  }

  // ---- write O: C-layout rows q = r (+8 for upper lanes), col d = lrow ----
  const int qrow0 = qb + w * 16 + ((lane & 16) ? 8 : 0);
#pragma unroll
  for (int r = 0; r < 8; ++r) {
    float* orow = Og + (size_t)(qrow0 + r) * D_HEAD + lrow;
    orow[0]  = acc0[r];
    orow[16] = acc1[r];
    orow[32] = acc2[r];
    orow[48] = acc3[r];
  }

  // ---- zero-fill causal tail of p_attn: keys >= qb+QB for rows [qb,qb+QB) ----
  const int tail0  = qb + QB;
  const int ntail4 = (S_LEN - tail0) >> 2;
  if (ntail4 > 0) {
    const v4f z = {0.f, 0.f, 0.f, 0.f};
    const int total = QB * ntail4;
    for (int i = tid; i < total; i += NTHREADS) {
      int row = i / ntail4, c4 = i % ntail4;
      __builtin_nontemporal_store(
          z, (v4f*)(Pg + (size_t)(qb + row) * S_LEN + tail0 + (c4 << 2)));
    }
  }
}

extern "C" void kernel_launch(void* const* d_in, const int* in_sizes, int n_in,
                              void* d_out, int out_size, void* d_ws, size_t ws_size,
                              hipStream_t stream) {
  const float* Q = (const float*)d_in[0];
  const float* K = (const float*)d_in[1];
  const float* V = (const float*)d_in[2];
  // d_in[3] is the causal mask; causality is applied analytically in-kernel.
  (void)in_sizes; (void)n_in; (void)d_ws; (void)ws_size; (void)out_size;

  const int B = 4, H = 16;
  float* O = (float*)d_out;                                   // [B,H,S,D]
  float* P = (float*)d_out + (size_t)B * H * S_LEN * D_HEAD;  // [B,H,S,S]

  const int grid = B * H * (S_LEN / QB);   // 1024 workgroups
  attn_fwd_causal<<<grid, NTHREADS, 0, stream>>>(Q, K, V, O, P);
}
